// QuantumFilter_65481071410722
// MI455X (gfx1250) — matrix operand order repaired and checked by reading, not compiled, not run on hardware
//
#include <hip/hip_runtime.h>
#include <math.h>

typedef __attribute__((ext_vector_type(16))) __bf16 v16bf;
typedef __attribute__((ext_vector_type(8)))  __bf16 v8bf;
typedef __attribute__((ext_vector_type(8)))  float  v8f;

#define BROWS 8192
#define NCOLS 784
#define KP    800      // K padded to 25 * 32
#define NKSTEP 25

// ---------------- helpers ----------------
__device__ __forceinline__ v8f zero8f() {
  v8f z;
#pragma unroll
  for (int i = 0; i < 8; ++i) z[i] = 0.f;
  return z;
}

// A fragment (16x32 bf16): lane holds M=lane&15; chunks K=[8h..8h+7] and [16+8h..16+8h+7]
// base pointer already includes row*KP + 8*(lane>>4)
__device__ __forceinline__ v16bf loadA(const __bf16* p, int k0) {
  union { v16bf v; v8bf h[2]; } u;
  u.h[0] = *(const v8bf*)(p + k0);
  u.h[1] = *(const v8bf*)(p + k0 + 16);
  return u.v;
}

// B fragment (32x16 bf16): lane holds N=lane&15; chunks K=[8h..8h+7] and [16+8h..16+8h+7]
// base pointer already includes col*KP + 8*(lane>>4)  (col clamped; no guard needed)
__device__ __forceinline__ v16bf loadB(const __bf16* p, int k0) {
  union { v16bf v; v8bf h[2]; } u;
  u.h[0] = *(const v8bf*)(p + k0);
  u.h[1] = *(const v8bf*)(p + k0 + 16);
  return u.v;
}

__device__ __forceinline__ v8f wmma_bf16(v16bf a, v16bf b, v8f c) {
  return __builtin_amdgcn_wmma_f32_16x16x32_bf16(false, a, false, b, (short)0, c,
                                                 false, false);
}

__device__ __forceinline__ void load_step(v16bf ah[2], v16bf al[2], v16bf bh[4],
                                          v16bf bl[4], const __bf16* const pAh[2],
                                          const __bf16* const pAl[2],
                                          const __bf16* const pBh[4],
                                          const __bf16* const pBl[4], int k0) {
#pragma unroll
  for (int mi = 0; mi < 2; ++mi) {
    ah[mi] = loadA(pAh[mi], k0);
    al[mi] = loadA(pAl[mi], k0);
  }
#pragma unroll
  for (int ni = 0; ni < 4; ++ni) {
    bh[ni] = loadB(pBh[ni], k0);
    bl[ni] = loadB(pBl[ni], k0);
  }
}

// 24 WMMAs: hi*hi + hi*lo + lo*hi per 16x16 tile (bf16x3 ~ fp32 accuracy)
__device__ __forceinline__ void do_wmma(v8f acc[2][4], const v16bf ah[2],
                                        const v16bf al[2], const v16bf bh[4],
                                        const v16bf bl[4]) {
#pragma unroll
  for (int mi = 0; mi < 2; ++mi)
#pragma unroll
    for (int ni = 0; ni < 4; ++ni) {
      acc[mi][ni] = wmma_bf16(ah[mi], bh[ni], acc[mi][ni]);
      acc[mi][ni] = wmma_bf16(ah[mi], bl[ni], acc[mi][ni]);
      acc[mi][ni] = wmma_bf16(al[mi], bh[ni], acc[mi][ni]);
    }
}

// ---------------- kernel 1: fp32 -> (hi,lo) bf16 split with K padding ----------------
__global__ void prep_split(const float* __restrict__ x, const float* __restrict__ w,
                           __bf16* __restrict__ xh, __bf16* __restrict__ xl,
                           __bf16* __restrict__ wh, __bf16* __restrict__ wl) {
  int row = blockIdx.x;
  const float* src;
  __bf16 *dh, *dl;
  int r;
  if (row < BROWS) { src = x; dh = xh; dl = xl; r = row; }
  else             { src = w; dh = wh; dl = wl; r = row - BROWS; }
  for (int c = threadIdx.x; c < KP; c += blockDim.x) {
    float v = (c < NCOLS) ? src[(size_t)r * NCOLS + c] : 0.f;
    __bf16 hi = (__bf16)v;
    __bf16 lo = (__bf16)(v - (float)hi);
    dh[(size_t)r * KP + c] = hi;
    dl[(size_t)r * KP + c] = lo;
  }
}

// ---------------- kernel 2: WMMA GEMM + quantum measurement + bias + BN stats ----------------
// block = 256 threads = 8 waves arranged 4(M) x 2(N); wave tile = 32x64; block tile = 128x128
__global__ __launch_bounds__(256) void gemm_fused(
    const __bf16* __restrict__ xh, const __bf16* __restrict__ xl,
    const __bf16* __restrict__ wh, const __bf16* __restrict__ wl,
    const float* __restrict__ x, const float* __restrict__ resb,
    float* __restrict__ outp, float* __restrict__ colsum, float* __restrict__ colsq) {
  const int lane = threadIdx.x & 31;
  const int wave = threadIdx.x >> 5;
  const int mw = wave & 3;       // 4 waves along M
  const int nw = wave >> 2;      // 2 waves along N
  const int m16 = lane & 15;
  const int half = lane >> 4;

  const int rowWave = blockIdx.y * 128 + mw * 32;   // 2 M-subtiles of 16
  const int colWave = blockIdx.x * 128 + nw * 64;   // 4 N-subtiles of 16

  v8f acc[2][4];
#pragma unroll
  for (int mi = 0; mi < 2; ++mi)
#pragma unroll
    for (int ni = 0; ni < 4; ++ni) acc[mi][ni] = zero8f();

  const __bf16* pAh[2];
  const __bf16* pAl[2];
#pragma unroll
  for (int mi = 0; mi < 2; ++mi) {
    size_t base = (size_t)(rowWave + mi * 16 + m16) * KP + 8 * half;
    pAh[mi] = xh + base;
    pAl[mi] = xl + base;
  }
  const __bf16* pBh[4];
  const __bf16* pBl[4];
#pragma unroll
  for (int ni = 0; ni < 4; ++ni) {
    int f = colWave + ni * 16 + m16;
    // clamp out-of-range columns to row 0: loads are safe, accumulator garbage for
    // those columns is never stored (epilogue and stats are guarded by f < NCOLS).
    size_t base = (size_t)((f < NCOLS) ? f : 0) * KP + 8 * half;
    pBh[ni] = wh + base;
    pBl[ni] = wl + base;
  }

  // 2-deep software pipeline: loads for step k+1 are issued before WMMAs of step k
  v16bf ah0[2], al0[2], bh0[4], bl0[4];
  v16bf ah1[2], al1[2], bh1[4], bl1[4];
  load_step(ah0, al0, bh0, bl0, pAh, pAl, pBh, pBl, 0);
  int k = 32;
#pragma unroll 1
  for (int it = 0; it < (NKSTEP - 1) / 2; ++it) {  // 12 iterations, steps 1..24
    __builtin_prefetch(pAh[0] + k + 64, 0, 1);
    __builtin_prefetch(pBh[0] + k + 64, 0, 1);
    load_step(ah1, al1, bh1, bl1, pAh, pAl, pBh, pBl, k);
    do_wmma(acc, ah0, al0, bh0, bl0);
    load_step(ah0, al0, bh0, bl0, pAh, pAl, pBh, pBl, k + 32);
    do_wmma(acc, ah1, al1, bh1, bl1);
    k += 64;
  }
  do_wmma(acc, ah0, al0, bh0, bl0);  // step 24

  // epilogue: + quantum measurement + bias, store pre-BN, accumulate column stats
#pragma unroll
  for (int ni = 0; ni < 4; ++ni) {
    const int f = colWave + ni * 16 + m16;
    const bool valid = (f < NCOLS);
    float s1 = 0.f, s2 = 0.f;
    // feature f = 4*p + q ; patch p = (pr, pc) ; meas_q = prod of cos over wires S_q
    const int q = f & 3;
    const int p = f >> 2;
    const int pr = p / 14;
    const int pc = p - pr * 14;
    const int pixoff = pr * 56 + pc * 2 + (q & 1);  // (2*pr)*28 + 2*pc (+1 for odd q)
    const float rb = valid ? resb[f] : 0.f;
#pragma unroll
    for (int mi = 0; mi < 2; ++mi) {
#pragma unroll
      for (int r = 0; r < 8; ++r) {
        const int b = rowWave + mi * 16 + 8 * half + r;  // C/D layout: M = r + 8*half
        if (valid) {
          const float* px = x + (size_t)b * NCOLS + pixoff;
          float m = __cosf(px[0]);
          if (q >= 2) m *= __cosf(px[28]);
          float v = acc[mi][ni][r] + m + rb;
          outp[(size_t)b * NCOLS + f] = v;
          s1 += v;
          s2 += v * v;
        }
      }
    }
    s1 += __shfl_xor(s1, 16, 32);  // lanes l and l+16 share column f
    s2 += __shfl_xor(s2, 16, 32);
    if (valid && half == 0) {
      atomicAdd(&colsum[f], s1);
      atomicAdd(&colsq[f], s2);
    }
  }
}

// ---------------- kernel 3: fold stats into scale/shift ----------------
__global__ void bn_finalize(const float* __restrict__ colsum, const float* __restrict__ colsq,
                            const float* __restrict__ gamma, const float* __restrict__ beta,
                            float* __restrict__ scale, float* __restrict__ shift) {
  int f = blockIdx.x * blockDim.x + threadIdx.x;
  if (f >= NCOLS) return;
  const float inv = 1.0f / (float)BROWS;
  float mean = colsum[f] * inv;
  float var = colsq[f] * inv - mean * mean;  // biased variance, matches reference
  float sc = gamma[f] * rsqrtf(var + 1e-5f);
  scale[f] = sc;
  shift[f] = beta[f] - mean * sc;
}

// ---------------- kernel 4: apply BN (float4 vectorized; 784 % 4 == 0) ----------------
__global__ void bn_apply(const float* __restrict__ outp, const float* __restrict__ scale,
                         const float* __restrict__ shift, float* __restrict__ out) {
  size_t i4 = ((size_t)blockIdx.x * blockDim.x + threadIdx.x) * 4;
  if (i4 >= (size_t)BROWS * NCOLS) return;
  int f = (int)(i4 % NCOLS);
  float4 v = *(const float4*)(outp + i4);
  float4 sc = *(const float4*)(scale + f);
  float4 sh = *(const float4*)(shift + f);
  float4 o;
  o.x = v.x * sc.x + sh.x;
  o.y = v.y * sc.y + sh.y;
  o.z = v.z * sc.z + sh.z;
  o.w = v.w * sc.w + sh.w;
  *(float4*)(out + i4) = o;
}

// ---------------- launcher ----------------
extern "C" void kernel_launch(void* const* d_in, const int* in_sizes, int n_in,
                              void* d_out, int out_size, void* d_ws, size_t ws_size,
                              hipStream_t stream) {
  const float* x     = (const float*)d_in[0];
  // d_in[1] = rz_angles: provably has no effect on the output (RZ phases vanish in
  // Z-basis probabilities through the CNOT-chain permutation) -> unused.
  const float* res_w = (const float*)d_in[2];
  const float* res_b = (const float*)d_in[3];
  const float* gamma = (const float*)d_in[4];
  const float* beta  = (const float*)d_in[5];
  float* out = (float*)d_out;

  char* ws = (char*)d_ws;
  size_t off = 0;
  auto take = [&](size_t bytes) -> void* {
    void* p = ws + off;
    off += (bytes + 255) & ~(size_t)255;
    return p;
  };
  __bf16* xh = (__bf16*)take((size_t)BROWS * KP * sizeof(__bf16));
  __bf16* xl = (__bf16*)take((size_t)BROWS * KP * sizeof(__bf16));
  __bf16* wh = (__bf16*)take((size_t)NCOLS * KP * sizeof(__bf16));
  __bf16* wl = (__bf16*)take((size_t)NCOLS * KP * sizeof(__bf16));
  float* outp  = (float*)take((size_t)BROWS * NCOLS * sizeof(float));
  float* stats = (float*)take((size_t)NCOLS * 2 * sizeof(float));
  float* scale = (float*)take((size_t)NCOLS * sizeof(float));
  float* shift = (float*)take((size_t)NCOLS * sizeof(float));
  float* colsum = stats;
  float* colsq  = stats + NCOLS;

  // 1) bf16 hi/lo split of x and res_w (K padded to 800 with zeros)
  prep_split<<<BROWS + NCOLS, 256, 0, stream>>>(x, res_w, xh, xl, wh, wl);

  // 2) zero column statistics (capturable async memset)
  hipMemsetAsync(stats, 0, (size_t)NCOLS * 2 * sizeof(float), stream);

  // 3) fused WMMA GEMM + measurement + bias + stats
  dim3 grid((NCOLS + 127) / 128, BROWS / 128);  // (7, 64)
  gemm_fused<<<grid, 256, 0, stream>>>(xh, xl, wh, wl, x, res_b, outp, colsum, colsq);

  // 4) BN scale/shift
  bn_finalize<<<(NCOLS + 255) / 256, 256, 0, stream>>>(colsum, colsq, gamma, beta,
                                                       scale, shift);

  // 5) apply BN -> d_out
  size_t n4 = (size_t)BROWS * NCOLS / 4;
  bn_apply<<<(unsigned)((n4 + 255) / 256), 256, 0, stream>>>(outp, scale, shift, out);
}
